// TDDA_20289425506637
// MI455X (gfx1250) — matrix operand order repaired and checked by reading, not compiled, not run on hardware
//
#include <hip/hip_runtime.h>
#include <math.h>

// ---------------------------------------------------------------------------
// MI455X (gfx1250) implementation of the temporal deformable-alignment net.
//
//  * The 128x1089 @ 1089x1089 MLP input is rank-1 (arange(t)/20 broadcast),
//    so it collapses to a row-sum of mlp_w  -> removes the biggest GEMM.
//  * All Cin=32 3x3 convs (deform GEMMs, offset convs, residual blocks) and
//    the 1x1 attention convs are implicit GEMMs with K = 9*32 or 32 and run
//    on V_WMMA_F32_16X16X32_F16 (f16 in, f32 accumulate).
//  * Activations live as zero-padded 35x35x32 HWC f16 frames (+16px dump
//    tail) so the WMMA A-fragment per lane is two contiguous 8-half loads
//    and ALL stores are branchless (select on address, no EXEC branching).
//  * ~50 GFLOP total, ~100-200MB traffic -> fully L2 (192MB) resident.
// ---------------------------------------------------------------------------

typedef __attribute__((ext_vector_type(16))) _Float16 v16h;
typedef __attribute__((ext_vector_type(8)))  _Float16 v8h;
typedef __attribute__((ext_vector_type(8)))  float    v8f;

#define HDI __device__ __forceinline__

namespace {
constexpr int Tt  = 128;
constexpr int Hh  = 33;
constexpr int Ww  = 33;
constexpr int HWp = Hh * Ww;            // 1089
constexpr int CH  = 32;
constexpr int PD  = 35;                 // padded dim
constexpr int PHW = PD * PD;            // 1225
constexpr int NT  = (HWp + 15) / 16;    // 69 pixel tiles of 16
constexpr int PT  = NT * 16;            // 1104 tile-padded pixel count
constexpr int FSX = (PHW + 16) * CH;    // HWC frame stride incl. dump tail
constexpr int Jf  = Tt - 4;             // 124 center frames
constexpr int OFS = CH * PT;            // offset-buffer frame stride (f32)
}

HDI int iminc(int a, int b) { return a < b ? a : b; }
HDI int imaxc(int a, int b) { return a > b ? a : b; }

// ------------------------------- fill ---------------------------------------
__global__ void k_fill0(unsigned int* p, size_t n) {
  size_t i = (size_t)blockIdx.x * blockDim.x + threadIdx.x;
  size_t stride = (size_t)gridDim.x * blockDim.x;
  for (; i < n; i += stride) p[i] = 0u;
}

// --------------------------- rowsum of mlp_w --------------------------------
__global__ void k_rowsum(const float* __restrict__ w, float* __restrict__ s) {
  __shared__ float sm[256];
  int j = blockIdx.x;
  float acc = 0.f;
  for (int k = threadIdx.x; k < HWp; k += blockDim.x) acc += w[(size_t)j * HWp + k];
  sm[threadIdx.x] = acc;
  __syncthreads();
  for (int st = 128; st > 0; st >>= 1) {
    if ((int)threadIdx.x < st) sm[threadIdx.x] += sm[threadIdx.x + st];
    __syncthreads();
  }
  if (threadIdx.x == 0) s[j] = sm[0];
}

// ----------------- pe = sigmoid((t/20)*rowsum + b); imgs = x*pe -------------
__global__ void k_pe_img(const float* __restrict__ x, const float* __restrict__ s,
                         const float* __restrict__ mlp_b, float* __restrict__ imgs,
                         float* __restrict__ out_aligned) {
  int idx = blockIdx.x * blockDim.x + threadIdx.x;
  if (idx >= Tt * HWp) return;
  int t = idx / HWp, p = idx % HWp;
  float z  = ((float)t * 0.05f) * s[p] + mlp_b[p];
  float pe = 1.f / (1.f + expf(-z));
  float v  = x[idx] * pe;
  imgs[idx] = v;
  if (t >= 2 && t < Tt - 2)                     // cen_raw -> aligned channel 2
    out_aligned[(size_t)(t - 2) * 5 * HWp + 2 * HWp + p] = v;
}

// ---------------- feats = relu(conv3x3(imgs, fe_w)), Cin=1 ------------------
__global__ void k_fe_conv(const float* __restrict__ imgs, const float* __restrict__ w,
                          const float* __restrict__ b, _Float16* __restrict__ feats) {
  int idx = blockIdx.x * blockDim.x + threadIdx.x;
  if (idx >= Tt * HWp) return;
  int t = idx / HWp, p = idx % HWp, y = p / Ww, x = p % Ww;
  float v[9];
#pragma unroll
  for (int k = 0; k < 9; k++) {
    int iy = y + k / 3 - 1, ix = x + k % 3 - 1;
    v[k] = (iy >= 0 && iy < Hh && ix >= 0 && ix < Ww)
               ? imgs[(size_t)t * HWp + iy * Ww + ix] : 0.f;
  }
  size_t base = (size_t)t * FSX + ((y + 1) * PD + (x + 1)) * CH;
#pragma unroll 4
  for (int co = 0; co < CH; co++) {
    float acc = b[co];
#pragma unroll
    for (int k = 0; k < 9; k++) acc += v[k] * w[co * 9 + k];
    feats[base + co] = (_Float16)(acc > 0.f ? acc : 0.f);
  }
}

// -------- weight transform: OIHW (cout<=32, 32, 3, 3) -> [9][2][16][32] -----
__global__ void k_xform3(const float* __restrict__ w, int cout, _Float16* __restrict__ wB) {
  int idx = blockIdx.x * blockDim.x + threadIdx.x;
  if (idx >= 9 * 2 * 16 * 32) return;
  int c = idx & 31, n = (idx >> 5) & 15, t = (idx >> 9) & 1, k = idx >> 10;
  int ch = t * 16 + n;
  float v = (ch < cout) ? w[((size_t)ch * 32 + c) * 9 + k] : 0.f;
  wB[idx] = (_Float16)v;
}

// -------- weight transform: (16,32,1,1) -> [16][32] f16 ---------------------
__global__ void k_xform1(const float* __restrict__ w, _Float16* __restrict__ wB) {
  int idx = blockIdx.x * blockDim.x + threadIdx.x;
  if (idx < 16 * 32) wB[idx] = (_Float16)w[idx];
}

// ------------------------- WMMA 1x1 conv (32 -> 16) -------------------------
// in: padded HWC f16 (frame-major), out: [frame][PT][16] f16 (tile-padded,
// so stores are unconditional)
__global__ void k_conv1x1_wmma(const _Float16* __restrict__ in,
                               const _Float16* __restrict__ wB,
                               const float* __restrict__ bias,
                               _Float16* __restrict__ out) {
  int wave = threadIdx.x >> 5, lane = threadIdx.x & 31;
  int tile = blockIdx.x * (blockDim.x >> 5) + wave;
  if (tile >= NT) return;
  int frame = blockIdx.y;
  int n = lane & 15, half = lane >> 4;
  int pm = iminc(tile * 16 + n, HWp - 1);
  int y = pm / Ww, x = pm % Ww;
  const _Float16* ap = in + (size_t)frame * FSX + ((y + 1) * PD + (x + 1)) * CH;
  __builtin_prefetch(ap, 0, 1);
  v8h alo = *(const v8h*)(ap + half * 16);
  v8h ahi = *(const v8h*)(ap + half * 16 + 8);
  v16h A;
#pragma unroll
  for (int e = 0; e < 8; e++) { A[e] = alo[e]; A[e + 8] = ahi[e]; }
  v16h B = *(const v16h*)(wB + n * 32 + half * 16);
  v8f acc;
  float bv = bias[n];
#pragma unroll
  for (int r = 0; r < 8; r++) acc[r] = bv;
  acc = __builtin_amdgcn_wmma_f32_16x16x32_f16(false, A, false, B, (short)0, acc,
                                               false, false);
  _Float16* of = out + (size_t)frame * PT * 16;
#pragma unroll
  for (int r = 0; r < 8; r++) {
    int p = tile * 16 + r + 8 * half;           // < PT, always in bounds
    of[(size_t)p * 16 + n] = (_Float16)acc[r];
  }
}

// --------------------- generic WMMA 3x3 conv (K = 288) ----------------------
// mode 0: out padded HWC f16 (optional relu, optional residual add)
// mode 1: out f32 [frame][32][PT] (tile/channel padded -> unconditional)
__global__ void k_conv3x3_wmma(const _Float16* __restrict__ in,
                               const _Float16* __restrict__ wB,   // [9][2][16][32]
                               const float* __restrict__ bias,    // nullable
                               int cout, int relu_flag,
                               const _Float16* __restrict__ res,  // nullable
                               _Float16* __restrict__ out_hwc,
                               float* __restrict__ out_pch, int mode) {
  int wave = threadIdx.x >> 5, lane = threadIdx.x & 31;
  int tile = blockIdx.x * (blockDim.x >> 5) + wave;
  if (tile >= NT) return;
  int frame = blockIdx.y;
  int n = lane & 15, half = lane >> 4;
  int pm = iminc(tile * 16 + n, HWp - 1);
  int y = pm / Ww, x = pm % Ww;
  const _Float16* inf = in + (size_t)frame * FSX;
  __builtin_prefetch(inf + (y * PD + x) * CH, 0, 1);
  float b0 = 0.f, b1 = 0.f;
  if (bias) {
    if (n < cout)      b0 = bias[n];
    if (16 + n < cout) b1 = bias[16 + n];
  }
  v8f acc0, acc1;
#pragma unroll
  for (int r = 0; r < 8; r++) { acc0[r] = b0; acc1[r] = b1; }
#pragma unroll
  for (int k = 0; k < 9; k++) {
    const _Float16* ap = inf + ((y + k / 3) * PD + (x + k % 3)) * CH;
    v8h alo = *(const v8h*)(ap + half * 16);
    v8h ahi = *(const v8h*)(ap + half * 16 + 8);
    v16h A;
#pragma unroll
    for (int e = 0; e < 8; e++) { A[e] = alo[e]; A[e + 8] = ahi[e]; }
    v16h B0 = *(const v16h*)(wB + ((k * 2 + 0) * 16 + n) * 32 + half * 16);
    v16h B1 = *(const v16h*)(wB + ((k * 2 + 1) * 16 + n) * 32 + half * 16);
    acc0 = __builtin_amdgcn_wmma_f32_16x16x32_f16(false, A, false, B0, (short)0,
                                                  acc0, false, false);
    acc1 = __builtin_amdgcn_wmma_f32_16x16x32_f16(false, A, false, B1, (short)0,
                                                  acc1, false, false);
  }
  if (mode == 0) {
    _Float16* of = out_hwc + (size_t)frame * FSX;
    const _Float16* rf = res ? res + (size_t)frame * FSX : nullptr;
#pragma unroll
    for (int r = 0; r < 8; r++) {
      int p = tile * 16 + r + 8 * half;
      int yy = p / Ww, xx = p % Ww;
      // branchless: out-of-image pixels land in the per-frame dump tail
      int base = (p < HWp) ? ((yy + 1) * PD + (xx + 1)) * CH : PHW * CH + r * CH;
      float v0 = acc0[r], v1 = acc1[r];
      if (rf) { v0 += (float)rf[base + n]; v1 += (float)rf[base + 16 + n]; }
      if (relu_flag) { v0 = v0 > 0.f ? v0 : 0.f; v1 = v1 > 0.f ? v1 : 0.f; }
      of[base + n]      = (_Float16)v0;
      of[base + 16 + n] = (_Float16)v1;
    }
  } else {
    float* of = out_pch + (size_t)frame * OFS;
#pragma unroll
    for (int r = 0; r < 8; r++) {
      int p = tile * 16 + r + 8 * half;       // < PT
      of[(size_t)n * PT + p]        = acc0[r];
      of[(size_t)(16 + n) * PT + p] = acc1[r];
    }
  }
}

// ------------------- deformable conv: bilinear sample + WMMA ----------------
// in: padded HWC f16 (frame-major, pre-offset), off: [J][32][PT] f32,
// wB: [9][2][16][32], out: padded HWC f16
__global__ void k_deform_wmma(const _Float16* __restrict__ in,
                              const float* __restrict__ off,
                              const _Float16* __restrict__ wB,
                              _Float16* __restrict__ out) {
  int wave = threadIdx.x >> 5, lane = threadIdx.x & 31;
  int tile = blockIdx.x * (blockDim.x >> 5) + wave;
  if (tile >= NT) return;
  int frame = blockIdx.y;
  int n = lane & 15, half = lane >> 4;
  int pm = iminc(tile * 16 + n, HWp - 1);
  int y = pm / Ww, x = pm % Ww;
  const _Float16* inf = in + (size_t)frame * FSX;
  __builtin_prefetch(inf + (y * PD + x) * CH, 0, 1);
  const float* offf = off + (size_t)frame * OFS;
  v8f acc0, acc1;
#pragma unroll
  for (int r = 0; r < 8; r++) { acc0[r] = 0.f; acc1[r] = 0.f; }
  for (int k = 0; k < 9; k++) {
    float dy = offf[(size_t)(2 * k + 0) * PT + pm];
    float dx = offf[(size_t)(2 * k + 1) * PT + pm];
    float py = (float)(y - 1 + k / 3) + dy;
    float px = (float)(x - 1 + k % 3) + dx;
    float fy = floorf(py), fx = floorf(px);
    float ly = py - fy, lx = px - fx;
    int y0 = (int)fy, x0 = (int)fx;
    float vbuf[16];
#pragma unroll
    for (int e = 0; e < 16; e++) vbuf[e] = 0.f;
#pragma unroll
    for (int cyi = 0; cyi < 2; cyi++) {
#pragma unroll
      for (int cxi = 0; cxi < 2; cxi++) {
        int cy = y0 + cyi, cx = x0 + cxi;
        float wgt = (cyi ? ly : 1.f - ly) * (cxi ? lx : 1.f - lx);
        bool val = (cy >= 0) && (cy < Hh) && (cx >= 0) && (cx < Ww);
        wgt = val ? wgt : 0.f;                  // mask (keeps EXEC uniform)
        int ccy = iminc(imaxc(cy, 0), Hh - 1);
        int ccx = iminc(imaxc(cx, 0), Ww - 1);
        const _Float16* p = inf + ((ccy + 1) * PD + (ccx + 1)) * CH;
        v8h clo = *(const v8h*)(p + half * 16);
        v8h chi = *(const v8h*)(p + half * 16 + 8);
#pragma unroll
        for (int e = 0; e < 8; e++) {
          vbuf[e]     += wgt * (float)clo[e];
          vbuf[e + 8] += wgt * (float)chi[e];
        }
      }
    }
    v16h A;
#pragma unroll
    for (int e = 0; e < 16; e++) A[e] = (_Float16)vbuf[e];
    v16h B0 = *(const v16h*)(wB + ((k * 2 + 0) * 16 + n) * 32 + half * 16);
    v16h B1 = *(const v16h*)(wB + ((k * 2 + 1) * 16 + n) * 32 + half * 16);
    acc0 = __builtin_amdgcn_wmma_f32_16x16x32_f16(false, A, false, B0, (short)0,
                                                  acc0, false, false);
    acc1 = __builtin_amdgcn_wmma_f32_16x16x32_f16(false, A, false, B1, (short)0,
                                                  acc1, false, false);
  }
  _Float16* of = out + (size_t)frame * FSX;
#pragma unroll
  for (int r = 0; r < 8; r++) {
    int p = tile * 16 + r + 8 * half;
    int yy = p / Ww, xx = p % Ww;
    int base = (p < HWp) ? ((yy + 1) * PD + (xx + 1)) * CH : PHW * CH + r * CH;
    of[base + n]      = (_Float16)acc0[r];
    of[base + 16 + n] = (_Float16)acc1[r];
  }
}

// ------------------ agg = [mean, max] over concat(a1c, a2) ------------------
__global__ void k_agg(const _Float16* __restrict__ a1c, const _Float16* __restrict__ a2,
                      float* __restrict__ agg) {
  int idx = blockIdx.x * blockDim.x + threadIdx.x;
  if (idx >= Jf * HWp) return;
  int j = idx / HWp, p = idx % HWp;
  const _Float16* pa = a1c + ((size_t)j * PT + p) * 16;
  const _Float16* pb = a2  + ((size_t)j * PT + p) * 16;
  float sum = 0.f, mx = -3.4e38f;
#pragma unroll
  for (int c = 0; c < 16; c++) {
    float va = (float)pa[c], vb = (float)pb[c];
    sum += va + vb;
    mx = va > mx ? va : mx;
    mx = vb > mx ? vb : mx;
  }
  agg[(size_t)j * 2 * HWp + p]       = sum * (1.f / 32.f);
  agg[(size_t)j * 2 * HWp + HWp + p] = mx;
}

// ------- sig = sigmoid(conv3x3(agg, sq_w)); fused = [a1c*s0, a2*s1] ---------
__global__ void k_sig_fuse(const _Float16* __restrict__ a1c, const _Float16* __restrict__ a2,
                           const float* __restrict__ agg, const float* __restrict__ sq_w,
                           const float* __restrict__ sq_b, _Float16* __restrict__ fused) {
  int idx = blockIdx.x * blockDim.x + threadIdx.x;
  if (idx >= Jf * HWp) return;
  int j = idx / HWp, p = idx % HWp, y = p / Ww, x = p % Ww;
  float sg[2];
#pragma unroll
  for (int oc = 0; oc < 2; oc++) {
    float acc = sq_b[oc];
#pragma unroll
    for (int ic = 0; ic < 2; ic++) {
#pragma unroll
      for (int k = 0; k < 9; k++) {
        int iy = y + k / 3 - 1, ix = x + k % 3 - 1;
        float v = (iy >= 0 && iy < Hh && ix >= 0 && ix < Ww)
                      ? agg[(size_t)j * 2 * HWp + ic * HWp + iy * Ww + ix] : 0.f;
        acc += v * sq_w[(oc * 2 + ic) * 9 + k];
      }
    }
    sg[oc] = 1.f / (1.f + expf(-acc));
  }
  size_t base = (size_t)j * FSX + ((y + 1) * PD + (x + 1)) * CH;
  const _Float16* pa = a1c + ((size_t)j * PT + p) * 16;
  const _Float16* pb = a2  + ((size_t)j * PT + p) * 16;
#pragma unroll
  for (int c = 0; c < 16; c++) {
    fused[base + c]      = (_Float16)((float)pa[c] * sg[0]);
    fused[base + 16 + c] = (_Float16)((float)pb[c] * sg[1]);
  }
}

// ---------------- Cout=1 3x3 conv from padded HWC (branch / tail) -----------
__global__ void k_conv_out1(const _Float16* __restrict__ in, const float* __restrict__ w,
                            const float* __restrict__ bias, float* __restrict__ out,
                            int frame_stride) {
  int idx = blockIdx.x * blockDim.x + threadIdx.x;
  if (idx >= Jf * HWp) return;
  int j = idx / HWp, p = idx % HWp, y = p / Ww, x = p % Ww;
  float acc = bias ? bias[0] : 0.f;
  const _Float16* inf = in + (size_t)j * FSX;
#pragma unroll
  for (int k = 0; k < 9; k++) {
    const _Float16* ap = inf + ((y + k / 3) * PD + (x + k % 3)) * CH;
#pragma unroll 8
    for (int c = 0; c < CH; c++) acc += (float)ap[c] * w[c * 9 + k];
  }
  out[(size_t)j * frame_stride + p] = acc;
}

// --------------------- tc0: relu(conv3x3(aligned 5ch -> 32)) ----------------
__global__ void k_tc0(const float* __restrict__ aligned, const float* __restrict__ w,
                      const float* __restrict__ b, _Float16* __restrict__ h) {
  int idx = blockIdx.x * blockDim.x + threadIdx.x;
  if (idx >= Jf * HWp) return;
  int j = idx / HWp, p = idx % HWp, y = p / Ww, x = p % Ww;
  float v[45];
#pragma unroll
  for (int ci = 0; ci < 5; ci++) {
#pragma unroll
    for (int k = 0; k < 9; k++) {
      int iy = y + k / 3 - 1, ix = x + k % 3 - 1;
      v[ci * 9 + k] = (iy >= 0 && iy < Hh && ix >= 0 && ix < Ww)
                          ? aligned[(size_t)j * 5 * HWp + ci * HWp + iy * Ww + ix] : 0.f;
    }
  }
  size_t base = (size_t)j * FSX + ((y + 1) * PD + (x + 1)) * CH;
#pragma unroll 4
  for (int co = 0; co < CH; co++) {
    float acc = b[co];
#pragma unroll
    for (int m = 0; m < 45; m++) acc += v[m] * w[co * 45 + m];
    h[base + co] = (_Float16)(acc > 0.f ? acc : 0.f);
  }
}

// ------------------------------- arange -------------------------------------
__global__ void k_iota(float* out) {
  int j = threadIdx.x;
  if (j < Jf) out[j] = (float)(j + 2);
}

// ---------------------------------------------------------------------------
extern "C" void kernel_launch(void* const* d_in, const int* in_sizes, int n_in,
                              void* d_out, int out_size, void* d_ws, size_t ws_size,
                              hipStream_t stream) {
  const float* x      = (const float*)d_in[0];
  const float* mlp_w  = (const float*)d_in[1];
  const float* mlp_b  = (const float*)d_in[2];
  const float* fe_w   = (const float*)d_in[3];
  const float* fe_b   = (const float*)d_in[4];
  const float* c0_w   = (const float*)d_in[5];
  const float* c0_b   = (const float*)d_in[6];
  const float* c1_w   = (const float*)d_in[7];
  const float* c1_b   = (const float*)d_in[8];
  const float* sq_w   = (const float*)d_in[9];
  const float* sq_b   = (const float*)d_in[10];
  const float* off1_w = (const float*)d_in[11];
  const float* off1_b = (const float*)d_in[12];
  const float* dw1    = (const float*)d_in[13];
  const float* off2_w = (const float*)d_in[14];
  const float* off2_b = (const float*)d_in[15];
  const float* dw2    = (const float*)d_in[16];
  const float* conv_w = (const float*)d_in[17];
  const float* conv_b = (const float*)d_in[18];
  const float* tc0_w  = (const float*)d_in[19];
  const float* tc0_b  = (const float*)d_in[20];
  const float* rb_w1  = (const float*)d_in[21];
  const float* rb_b1  = (const float*)d_in[22];
  const float* rb_w2  = (const float*)d_in[23];
  const float* rb_b2  = (const float*)d_in[24];
  const float* tail_w = (const float*)d_in[25];

  float* out_final   = (float*)d_out;                    // [124][1089]
  float* out_idx     = out_final + (size_t)Jf * HWp;     // [124]
  float* out_aligned = out_idx + Jf;                     // [124][5][1089]

  // ---- workspace bump allocator (assumes ws_size ~>= 98MB) ----
  char* ws = (char*)d_ws;
  size_t off = 0;
  auto alloc = [&](size_t bytes) -> char* {
    char* p = ws + off;
    off += (bytes + 255) & ~(size_t)255;
    return p;
  };
  float*    s_row  = (float*)alloc(HWp * 4);
  float*    imgs   = (float*)alloc((size_t)Tt * HWp * 4);
  _Float16* feats  = (_Float16*)alloc((size_t)Tt * FSX * 2);
  _Float16* a1c    = (_Float16*)alloc((size_t)Jf * PT * 16 * 2);
  _Float16* a2b    = (_Float16*)alloc((size_t)Jf * PT * 16 * 2);
  float*    aggb   = (float*)alloc((size_t)Jf * 2 * HWp * 4);
  _Float16* fusedb = (_Float16*)alloc((size_t)Jf * FSX * 2);
  float*    offb   = (float*)alloc((size_t)Jf * OFS * 4);
  _Float16* al1    = (_Float16*)alloc((size_t)Jf * FSX * 2);
  _Float16* al2    = (_Float16*)alloc((size_t)Jf * FSX * 2);
  _Float16* hA     = (_Float16*)alloc((size_t)Jf * FSX * 2);
  _Float16* hB     = (_Float16*)alloc((size_t)Jf * FSX * 2);
  _Float16* tmp16  = (_Float16*)alloc((size_t)Jf * FSX * 2);
  _Float16* c0B    = (_Float16*)alloc(16 * 32 * 2);
  _Float16* c1B    = (_Float16*)alloc(16 * 32 * 2);
  _Float16* off1B  = (_Float16*)alloc(9216 * 2);
  _Float16* off2B  = (_Float16*)alloc(9216 * 2);
  _Float16* dw1B   = (_Float16*)alloc(9216 * 2);
  _Float16* dw2B   = (_Float16*)alloc(9216 * 2);
  _Float16* rb1B   = (_Float16*)alloc(5 * 9216 * 2);
  _Float16* rb2B   = (_Float16*)alloc(5 * 9216 * 2);
  size_t total_bytes = off;

  // ---- zero workspace (padded borders must be 0 for all HWC frames) ----
  {
    size_t nwords  = total_bytes / 4;
    size_t nblocks = (nwords + 255) / 256;
    int blocks = nblocks > 8192 ? 8192 : (int)nblocks;
    k_fill0<<<blocks, 256, 0, stream>>>((unsigned int*)d_ws, nwords);
  }

  const int NPIX = Jf * HWp;
  const int PB   = (NPIX + 255) / 256;
  const int TB   = (Tt * HWp + 255) / 256;
  dim3 gw((NT + 7) / 8, Jf);   // 8 waves (tiles) per 256-thread block

  // 1) rank-1 MLP collapse + positional encoding + imgs (+cen_raw to output)
  k_rowsum<<<HWp, 256, 0, stream>>>(mlp_w, s_row);
  k_pe_img<<<TB, 256, 0, stream>>>(x, s_row, mlp_b, imgs, out_aligned);

  // 2) feature extraction (Cin=1 -> VALU) into padded HWC f16
  k_fe_conv<<<TB, 256, 0, stream>>>(imgs, fe_w, fe_b, feats);

  // 3) weight transforms for all WMMA convs
  k_xform1<<<2, 256, 0, stream>>>(c0_w, c0B);
  k_xform1<<<2, 256, 0, stream>>>(c1_w, c1B);
  k_xform3<<<36, 256, 0, stream>>>(off1_w, 18, off1B);
  k_xform3<<<36, 256, 0, stream>>>(off2_w, 18, off2B);
  k_xform3<<<36, 256, 0, stream>>>(dw1, 32, dw1B);
  k_xform3<<<36, 256, 0, stream>>>(dw2, 32, dw2B);
  for (int l = 0; l < 5; l++) {
    k_xform3<<<36, 256, 0, stream>>>(rb_w1 + (size_t)l * 32 * 32 * 9, 32, rb1B + l * 9216);
    k_xform3<<<36, 256, 0, stream>>>(rb_w2 + (size_t)l * 32 * 32 * 9, 32, rb2B + l * 9216);
  }

  // 4) a1c (center 1x1 attention conv) once
  k_conv1x1_wmma<<<gw, 256, 0, stream>>>(feats + (size_t)2 * FSX, c0B, c0_b, a1c);

  // 5) four alignment branches (sequential, buffers reused)
  const int offs[4]  = {-2, -1, 1, 2};
  const int chmap[4] = {0, 1, 3, 4};
  for (int bi = 0; bi < 4; bi++) {
    const _Float16* neig = feats + (size_t)(2 + offs[bi]) * FSX;
    k_conv1x1_wmma<<<gw, 256, 0, stream>>>(neig, c1B, c1_b, a2b);
    k_agg<<<PB, 256, 0, stream>>>(a1c, a2b, aggb);
    k_sig_fuse<<<PB, 256, 0, stream>>>(a1c, a2b, aggb, sq_w, sq_b, fusedb);
    // off1 = conv3x3(fused) -> [J][32][PT] f32 (18 real channels)
    k_conv3x3_wmma<<<gw, 256, 0, stream>>>(fusedb, off1B, off1_b, 18, 0, nullptr,
                                           nullptr, offb, 1);
    // al1 = deform(neig, off1, dw1)
    k_deform_wmma<<<gw, 256, 0, stream>>>(neig, offb, dw1B, al1);
    // off2 = conv3x3(al1) -> [J][32][PT] f32
    k_conv3x3_wmma<<<gw, 256, 0, stream>>>(al1, off2B, off2_b, 18, 0, nullptr,
                                           nullptr, offb, 1);
    // al2 = deform(al1, off2, dw2)
    k_deform_wmma<<<gw, 256, 0, stream>>>(al1, offb, dw2B, al2);
    // branch output (Cout=1) -> aligned channel chmap[bi]
    k_conv_out1<<<PB, 256, 0, stream>>>(al2, conv_w, conv_b,
                                        out_aligned + (size_t)chmap[bi] * HWp, 5 * HWp);
  }

  // 6) temporal fusion (5 -> 32, VALU) into padded HWC f16
  k_tc0<<<PB, 256, 0, stream>>>(out_aligned, tc0_w, tc0_b, hA);

  // 7) five residual blocks (WMMA conv pairs, ping-pong hA/hB)
  for (int l = 0; l < 5; l++) {
    k_conv3x3_wmma<<<gw, 256, 0, stream>>>(hA, rb1B + l * 9216, rb_b1 + l * 32, 32, 1,
                                           nullptr, tmp16, nullptr, 0);
    k_conv3x3_wmma<<<gw, 256, 0, stream>>>(tmp16, rb2B + l * 9216, rb_b2 + l * 32, 32, 0,
                                           hA, hB, nullptr, 0);
    _Float16* t = hA; hA = hB; hB = t;
  }

  // 8) tail (Cout=1, no bias) -> final
  k_conv_out1<<<PB, 256, 0, stream>>>(hA, tail_w, nullptr, out_final, HWp);

  // 9) arange(2, t-2) output
  k_iota<<<1, 128, 0, stream>>>(out_idx);

  (void)in_sizes; (void)n_in; (void)out_size; (void)ws_size;
}
